// LabelClassifier_43576738185611
// MI455X (gfx1250) — matrix operand order repaired and checked by reading, not compiled, not run on hardware
//
#include <hip/hip_runtime.h>
#include <hip/hip_bf16.h>

// ---------------- problem constants ----------------
#define S_DIM 2048
#define B_DIM 16
#define H2_DIM 1024
#define H_DIM 512
#define N_SPANS 4096
#define T_TAGS 8192
#define L_DIM 256
#define V_DIM 50000
// padded V so every tile is rectangular; masked in stats
#define VSLICES 8
#define VCHUNK 64
#define CHUNKS_PER_SLICE 98           // 98*64 = 6272 cols per slice
#define SLICE_COLS (CHUNKS_PER_SLICE * VCHUNK)
#define V_PAD (VSLICES * SLICE_COLS)  // 50176
#define NEG_BIG (-3.0e38f)

typedef unsigned short BF;  // raw bf16 storage
typedef __attribute__((ext_vector_type(16))) __bf16 v16bf;
typedef __attribute__((ext_vector_type(8)))  float  v8f;

static __device__ __forceinline__ BF f2bf(float x) {
  unsigned int u = __builtin_bit_cast(unsigned int, x);
  u += 0x7FFFu + ((u >> 16) & 1u);  // round to nearest even
  return (BF)(u >> 16);
}

static __device__ __forceinline__ v16bf load_frag(const BF* p0, const BF* p1) {
  v16bf r;
  ((uint4*)&r)[0] = *(const uint4*)p0;   // elements 0..7
  ((uint4*)&r)[1] = *(const uint4*)p1;   // elements 8..15
  return r;
}

static __device__ __forceinline__ v8f wmma_bf16(v16bf a, v16bf b, v8f c) {
  return __builtin_amdgcn_wmma_f32_16x16x32_bf16(false, a, false, b, (short)0, c,
                                                 false, false);
}

// async global -> LDS copy of 16 bytes per lane (ASYNCcnt-tracked, no VGPR staging)
static __device__ __forceinline__ void async_copy_b128(unsigned lds_off,
                                                       const void* gaddr) {
  asm volatile("global_load_async_to_lds_b128 %0, %1, off"
               :
               : "v"(lds_off), "v"(gaddr)
               : "memory");
}
static __device__ __forceinline__ void wait_async0() {
  asm volatile("s_wait_asynccnt 0x0" ::: "memory");
}

// ---------------- conversion kernels ----------------
__global__ __launch_bounds__(256) void cvt_bf16_kernel(const float* __restrict__ src,
                                                       BF* __restrict__ dst, int n) {
  for (int i = blockIdx.x * blockDim.x + threadIdx.x; i < n; i += gridDim.x * blockDim.x)
    dst[i] = f2bf(src[i]);
}

__global__ __launch_bounds__(256) void pad_w2_kernel(const float* __restrict__ W2,
                                                     BF* __restrict__ dst) {
  const int total = V_PAD * L_DIM;
  for (int i = blockIdx.x * blockDim.x + threadIdx.x; i < total; i += gridDim.x * blockDim.x) {
    int row = i >> 8, col = i & 255;
    float v = (row < V_DIM) ? W2[(size_t)row * L_DIM + col] : 0.0f;
    dst[i] = f2bf(v);
  }
}

// ---------------- span gather: feat (N,1024) bf16 ----------------
__global__ __launch_bounds__(256) void gather_kernel(const float* __restrict__ hidden,
                                                     const int* __restrict__ span_bid,
                                                     const int* __restrict__ span_begin,
                                                     const int* __restrict__ span_end,
                                                     BF* __restrict__ featb) {
  int n = blockIdx.x;
  int bid = span_bid[n];
  int fidx = (span_begin[n] - 1 + S_DIM) % S_DIM;  // python modulo
  int eidx = span_end[n];
  const size_t fbase = ((size_t)fidx * B_DIM + bid) * H2_DIM;
  const size_t ebase = ((size_t)eidx * B_DIM + bid) * H2_DIM;
#pragma unroll
  for (int i = 0; i < 4; ++i) {
    int j = threadIdx.x + 256 * i;
    float v = (j < H_DIM) ? hidden[fbase + j] : hidden[ebase + j];
    featb[(size_t)n * H2_DIM + j] = f2bf(v);
  }
}

// ---------------- GEMM1: x = tanh(feat @ W1^T + b1) ----------------
// grid.x = N/16 row tiles, 256 threads (8 waves); wave w covers N-tiles 2w, 2w+1.
__global__ __launch_bounds__(256) void gemm1_kernel(const BF* __restrict__ featb,
                                                    const BF* __restrict__ w1b,
                                                    const float* __restrict__ b1,
                                                    float* __restrict__ xf,
                                                    BF* __restrict__ xb) {
  __shared__ __align__(16) BF sA[16 * H2_DIM];  // 32 KB feat strip
  const int mbase = blockIdx.x * 16;
  {
    const uint4* gs = (const uint4*)(featb + (size_t)mbase * H2_DIM);
    unsigned lbase = (unsigned)(uintptr_t)(&sA[0]);
    for (int i = threadIdx.x; i < 16 * H2_DIM / 8; i += 256)
      async_copy_b128(lbase + i * 16, gs + i);
  }
  wait_async0();
  __syncthreads();

  const int lane = threadIdx.x & 31, wave = threadIdx.x >> 5;
  const int col16 = lane & 15, hi = lane >> 4;

  v8f acc[2];
  acc[0] = (v8f){0.f, 0.f, 0.f, 0.f, 0.f, 0.f, 0.f, 0.f};
  acc[1] = acc[0];

  for (int kt = 0; kt < H2_DIM / 32; ++kt) {
    const int k0 = kt * 32;
    const BF* ap = sA + col16 * H2_DIM + k0 + 8 * hi;
    v16bf a = load_frag(ap, ap + 16);
#pragma unroll
    for (int t = 0; t < 2; ++t) {
      const int n0 = (wave * 2 + t) * 16;
      const BF* bp = w1b + (size_t)(n0 + col16) * H2_DIM + k0 + 8 * hi;
      v16bf b = load_frag(bp, bp + 16);
      acc[t] = wmma_bf16(a, b, acc[t]);
    }
  }
#pragma unroll
  for (int t = 0; t < 2; ++t) {
    const int col = (wave * 2 + t) * 16 + col16;
    const float bias = b1[col];
#pragma unroll
    for (int r = 0; r < 8; ++r) {
      const int row = mbase + r + 8 * hi;
      float v = tanhf(acc[t][r] + bias);
      xf[(size_t)row * L_DIM + col] = v;
      xb[(size_t)row * L_DIM + col] = f2bf(v);
    }
  }
}

// ---------------- GEMM2 fused with online softmax stats ----------------
// grid = (32 M-groups, 8 V-slices); 8 waves/WG, wave owns a 16-row strip.
// W2 chunks double-buffered in LDS via async-to-LDS loads (ASYNCcnt).
__global__ __launch_bounds__(256) void gemm2_stats_kernel(const BF* __restrict__ xb,
                                                          const BF* __restrict__ w2b,
                                                          const float* __restrict__ b2,
                                                          float* __restrict__ stats) {
  __shared__ __align__(16) BF sB[2][VCHUNK * L_DIM];  // 2 x 32 KB
  const int mgroup = blockIdx.x, vslice = blockIdx.y;
  const int lane = threadIdx.x & 31, wave = threadIdx.x >> 5;
  const int col16 = lane & 15, hi = lane >> 4;
  const int mbase = mgroup * 128 + wave * 16;

  // resident A fragments: 16 rows x 256 bf16 (64 VGPRs)
  v16bf aw[8];
  {
    const int arow = mbase + col16;
#pragma unroll
    for (int kt = 0; kt < 8; ++kt) {
      const BF* ap = xb + (size_t)arow * L_DIM + kt * 32 + 8 * hi;
      aw[kt] = load_frag(ap, ap + 16);
    }
  }

  // per-lane online softmax state for the 8 (M = r + 8*hi) rows
  float lm[8], ls[8];
#pragma unroll
  for (int r = 0; r < 8; ++r) { lm[r] = NEG_BIG; ls[r] = 0.f; }

  // kick off chunk 0 into buffer 0
  {
    const uint4* gs = (const uint4*)(w2b + (size_t)(vslice * SLICE_COLS) * L_DIM);
    unsigned lbase = (unsigned)(uintptr_t)(&sB[0][0]);
    for (int i = threadIdx.x; i < VCHUNK * L_DIM / 8; i += 256)
      async_copy_b128(lbase + i * 16, gs + i);
  }

  for (int c = 0; c < CHUNKS_PER_SLICE; ++c) {
    const int vbase = vslice * SLICE_COLS + c * VCHUNK;
    wait_async0();      // this wave's pending chunk done
    __syncthreads();    // chunk c resident; previous buffer free for reuse

    if (c + 1 < CHUNKS_PER_SLICE) {  // prefetch chunk c+1 into the other buffer
      const uint4* gs = (const uint4*)(w2b + (size_t)(vbase + VCHUNK) * L_DIM);
      unsigned lbase = (unsigned)(uintptr_t)(&sB[(c + 1) & 1][0]);
      for (int i = threadIdx.x; i < VCHUNK * L_DIM / 8; i += 256)
        async_copy_b128(lbase + i * 16, gs + i);
    }

    const BF* bbuf = &sB[c & 1][0];
#pragma unroll
    for (int nt = 0; nt < 4; ++nt) {
      const int ncol = vbase + nt * 16 + col16;
      v8f acc = (v8f){0.f, 0.f, 0.f, 0.f, 0.f, 0.f, 0.f, 0.f};
#pragma unroll
      for (int kt = 0; kt < 8; ++kt) {
        const BF* bp = bbuf + (nt * 16 + col16) * L_DIM + kt * 32 + 8 * hi;
        v16bf b = load_frag(bp, bp + 16);
        acc = wmma_bf16(aw[kt], b, acc);
      }
      const bool valid = (ncol < V_DIM);
      const float bias = valid ? b2[ncol] : 0.f;
#pragma unroll
      for (int r = 0; r < 8; ++r) {
        const float val = acc[r] + bias;
        const float d = valid ? (val - lm[r]) : NEG_BIG;   // -BIG => contributes 0
        const float e = __expf(-fabsf(d));                 // single exp per element
        ls[r] = (d <= 0.f) ? (ls[r] + e) : fmaf(ls[r], e, 1.0f);
        lm[r] = valid ? fmaxf(lm[r], val) : lm[r];
      }
    }
  }

  // butterfly merge of (m,s) across the 16 lanes of each half
#pragma unroll
  for (int r = 0; r < 8; ++r) {
#pragma unroll
    for (int off = 1; off < 16; off <<= 1) {
      const float om = __shfl_xor(lm[r], off, 32);
      const float os = __shfl_xor(ls[r], off, 32);
      const float nm = fmaxf(lm[r], om);
      ls[r] = ls[r] * __expf(lm[r] - nm) + os * __expf(om - nm);
      lm[r] = nm;
    }
  }
  if (col16 == 0) {
#pragma unroll
    for (int r = 0; r < 8; ++r) {
      const int row = mbase + r + 8 * hi;
      stats[((size_t)row * VSLICES + vslice) * 2 + 0] = lm[r];
      stats[((size_t)row * VSLICES + vslice) * 2 + 1] = ls[r];
    }
  }
}

// ---------------- merge per-slice stats -> logZ ----------------
__global__ __launch_bounds__(256) void merge_kernel(const float* __restrict__ stats,
                                                    float* __restrict__ logZ) {
  const int n = blockIdx.x * blockDim.x + threadIdx.x;
  if (n >= N_SPANS) return;
  float m = NEG_BIG;
#pragma unroll
  for (int i = 0; i < VSLICES; ++i) m = fmaxf(m, stats[((size_t)n * VSLICES + i) * 2]);
  float s = 0.f;
#pragma unroll
  for (int i = 0; i < VSLICES; ++i)
    s += stats[((size_t)n * VSLICES + i) * 2 + 1] *
         __expf(stats[((size_t)n * VSLICES + i) * 2] - m);
  logZ[n] = m + logf(s);
}

// ---------------- positional logits + focal term (1 wave / tag) ----------------
__global__ __launch_bounds__(256) void pos_kernel(const float* __restrict__ xf,
                                                  const float* __restrict__ W2,
                                                  const float* __restrict__ b2,
                                                  const int* __restrict__ tsid,
                                                  const int* __restrict__ tags,
                                                  const float* __restrict__ logZ,
                                                  float* __restrict__ per_t) {
  const int wave = threadIdx.x >> 5, lane = threadIdx.x & 31;
  const int t = blockIdx.x * 8 + wave;
  if (t >= T_TAGS) return;
  const int n = tsid[t], v = tags[t];
  const float* xr = xf + (size_t)n * L_DIM;
  const float* wr = W2 + (size_t)v * L_DIM;
  float p = 0.f;
#pragma unroll
  for (int i = 0; i < 8; ++i) {
    const int k = lane + 32 * i;
    p = fmaf(xr[k], wr[k], p);
  }
#pragma unroll
  for (int off = 16; off > 0; off >>= 1) p += __shfl_xor(p, off, 32);
  if (lane == 0) {
    const float lp = (p + b2[v]) - logZ[n];
    const float pos = __expf(lp);
    per_t[t] = -(1.0f - pos) * lp;  // GAMMA == 1
  }
}

// ---------------- deterministic final reduction ----------------
__global__ __launch_bounds__(256) void reduce_kernel(const float* __restrict__ per_t,
                                                     float* __restrict__ out) {
  __shared__ float sm[256];
  float s = 0.f;
#pragma unroll
  for (int i = 0; i < T_TAGS / 256; ++i) s += per_t[threadIdx.x + 256 * i];
  sm[threadIdx.x] = s;
  __syncthreads();
  for (int st = 128; st > 0; st >>= 1) {
    if ((int)threadIdx.x < st) sm[threadIdx.x] += sm[threadIdx.x + st];
    __syncthreads();
  }
  if (threadIdx.x == 0) out[0] = sm[0] / (8192.0f + 1e-5f);
}

// ---------------- launcher ----------------
extern "C" void kernel_launch(void* const* d_in, const int* in_sizes, int n_in,
                              void* d_out, int out_size, void* d_ws, size_t ws_size,
                              hipStream_t stream) {
  const float* hidden     = (const float*)d_in[0];
  const float* W1         = (const float*)d_in[1];
  const float* b1         = (const float*)d_in[2];
  const float* W2         = (const float*)d_in[3];
  const float* b2         = (const float*)d_in[4];
  const int*   span_bid   = (const int*)d_in[5];
  const int*   span_begin = (const int*)d_in[6];
  const int*   span_end   = (const int*)d_in[7];
  const int*   tag_span   = (const int*)d_in[8];
  const int*   tags       = (const int*)d_in[9];
  float*       out        = (float*)d_out;

  // workspace layout (all offsets 256B-aligned by construction)
  char* ws = (char*)d_ws;
  size_t off = 0;
  BF*    w1b   = (BF*)(ws + off); off += (size_t)L_DIM * H2_DIM * 2;       // 0.5 MB
  BF*    w2b   = (BF*)(ws + off); off += (size_t)V_PAD * L_DIM * 2;        // 25.7 MB
  BF*    featb = (BF*)(ws + off); off += (size_t)N_SPANS * H2_DIM * 2;     // 8 MB
  BF*    xb    = (BF*)(ws + off); off += (size_t)N_SPANS * L_DIM * 2;      // 2 MB
  float* xf    = (float*)(ws + off); off += (size_t)N_SPANS * L_DIM * 4;   // 4 MB
  float* stats = (float*)(ws + off); off += (size_t)N_SPANS * VSLICES * 2 * 4;
  float* logZ  = (float*)(ws + off); off += (size_t)N_SPANS * 4;
  float* per_t = (float*)(ws + off); off += (size_t)T_TAGS * 4;
  (void)ws_size; (void)in_sizes; (void)n_in; (void)out_size;

  cvt_bf16_kernel<<<256, 256, 0, stream>>>(W1, w1b, L_DIM * H2_DIM);
  pad_w2_kernel<<<2048, 256, 0, stream>>>(W2, w2b);
  gather_kernel<<<N_SPANS, 256, 0, stream>>>(hidden, span_bid, span_begin, span_end, featb);
  gemm1_kernel<<<N_SPANS / 16, 256, 0, stream>>>(featb, w1b, b1, xf, xb);
  gemm2_stats_kernel<<<dim3(N_SPANS / 128, VSLICES), 256, 0, stream>>>(xb, w2b, b2, stats);
  merge_kernel<<<N_SPANS / 256, 256, 0, stream>>>(stats, logZ);
  pos_kernel<<<T_TAGS / 8, 256, 0, stream>>>(xf, W2, b2, tag_span, tags, logZ, per_t);
  reduce_kernel<<<1, 256, 0, stream>>>(per_t, out);
}
